// RNNRecurrentEntitiyDecoder_58231166599771
// MI455X (gfx1250) — matrix operand order repaired and checked by reading, not compiled, not run on hardware
//
#include <hip/hip_runtime.h>
#include <hip/hip_bf16.h>

// Problem constants (from reference): B=128, S=128, K=64, D=256
#define BB 128
#define SS 128
#define KK 64
#define DD 256

typedef __attribute__((ext_vector_type(16))) __bf16 v16bf;
typedef __attribute__((ext_vector_type(8)))  __bf16 v8bf;
typedef __attribute__((ext_vector_type(8)))  float  v8f;

typedef unsigned int u32;
typedef __attribute__((ext_vector_type(4))) u32 u32x4;
typedef __attribute__((ext_vector_type(8))) int i32x8;
typedef __attribute__((ext_vector_type(4))) int i32x4;

#define HSTR 260   // float LDS row stride (conflict-free for column-strided access)
#define HBST 264   // bf16 LDS row stride for h_bf (16B-aligned rows, conflict-free)

// ---------------------------------------------------------------------------
// TDM: 1-row (nDwords x 1) tensor load from global to LDS.
// D# packed per CDNA5 ISA 8.3/8.4: group0 = {count|flags, lds_addr,
// global_addr[31:0], global_addr[56:32]|type=2}; group1 = {data_size=4B,
// tensor_dim0, tensor_dim1=1, tile_dim0, tile_dim1=1, dim0_stride, 0, 0}.
// This toolchain uses the 6-arg builtin: (g0, g1, g2, g3, g4, cpol).
// ---------------------------------------------------------------------------
__device__ __forceinline__ u32 lds_offset(const void* p) {
  return (u32)(uintptr_t)p;   // low 32 bits of flat shared addr = LDS offset
}

__device__ __forceinline__ void tdm_load_row(u32 lds_off, const void* gptr, u32 nDwords) {
  unsigned long long ga = (unsigned long long)(uintptr_t)gptr;
  u32x4 g0;
  g0[0] = 1u;                                            // count=1 (user D#)
  g0[1] = lds_off;                                       // lds_addr (bytes)
  g0[2] = (u32)(ga & 0xffffffffu);                       // global_addr[31:0]
  g0[3] = (u32)((ga >> 32) & 0x01ffffffu) | (2u << 30);  // addr[56:32] | type=2
  i32x8 g1;
  g1[0] = (int)(2u << 16);                               // data_size = 4 bytes
  g1[1] = (int)((nDwords & 0xffffu) << 16);              // tensor_dim0[15:0]
  g1[2] = (int)(1u << 16);                               // tensor_dim1 = 1
  g1[3] = (int)((nDwords & 0xffffu) << 16);              // tile_dim0
  g1[4] = 1;                                             // tile_dim1 = 1
  g1[5] = (int)nDwords;                                  // tensor_dim0_stride
  g1[6] = 0;
  g1[7] = 0;
  i32x4 z4 = {0, 0, 0, 0};
  i32x8 z8 = {0, 0, 0, 0, 0, 0, 0, 0};
  __builtin_amdgcn_tensor_load_to_lds(g0, g1, z4, z4, z8, 0);
}

// ---------------------------------------------------------------------------
// Generic Y = X[R,256] @ M[256,256] (fp32 in, bf16 WMMA, fp32 out). One-shot
// precompute for enc@W and keys@V; 8 waves/block, wave w owns cols [32w,32w+32).
// ---------------------------------------------------------------------------
__global__ __launch_bounds__(256) void gemm256_bf16(const float* __restrict__ X,
                                                    const float* __restrict__ M,
                                                    float* __restrict__ Y) {
  const int tid  = threadIdx.x;
  const int lane = tid & 31;
  const int wave = tid >> 5;
  const int half = lane >> 4;
  const int nlo  = lane & 15;
  const int row0 = blockIdx.x * 16;
  const int n0   = wave * 32;

  v8f acc0 = {}, acc1 = {};
  for (int kk = 0; kk < DD; kk += 32) {
    v16bf a, b0, b1;
#pragma unroll
    for (int e = 0; e < 16; ++e) {
      const int ka = kk + 8 * half + e + ((e & 8) ? 8 : 0);   // A: 16-bit 16x32 layout
      const int kb = kk + 8 * half + e + ((e & 8) ? 8 : 0);   // B: 16-bit 32x16 layout
      a[e]  = (__bf16)X[(size_t)(row0 + nlo) * DD + ka];
      b0[e] = (__bf16)M[(size_t)kb * DD + n0 + nlo];
      b1[e] = (__bf16)M[(size_t)kb * DD + n0 + 16 + nlo];
    }
    acc0 = __builtin_amdgcn_wmma_f32_16x16x32_bf16(false, a, false, b0, (short)0, acc0, false, false);
    acc1 = __builtin_amdgcn_wmma_f32_16x16x32_bf16(false, a, false, b1, (short)0, acc1, false, false);
  }
#pragma unroll
  for (int r = 0; r < 8; ++r) {
    const int m = r + 8 * half;
    const int n = n0 + nlo;
    Y[(size_t)(row0 + m) * DD + n]      = acc0[r];
    Y[(size_t)(row0 + m) * DD + n + 16] = acc1[r];
  }
}

// ---------------------------------------------------------------------------
// Recurrent scan: one WG owns 16 entities of one batch element for all 128
// steps. U pre-swizzled to B-fragment order in LDS (bf16, 128 KB); h kept in
// LDS as fp32 (gates/holds/norm) + bf16 row-major (vectorized A fragments).
// enc_t / encW_t streamed by the Tensor Data Mover, double buffered.
// ---------------------------------------------------------------------------
__global__ __launch_bounds__(256) void entity_scan(
    const float* __restrict__ enc,          // [B,S,D]
    const unsigned char* __restrict__ mask, // [B,S] (bool)
    const float* __restrict__ keys,         // [B,K,D]
    const float* __restrict__ U,            // [D,D]
    const float* __restrict__ encW,         // [B,S,D]  (precomputed enc @ W)
    const float* __restrict__ keysV,        // [B,K,D]  (precomputed keys @ V)
    float* __restrict__ out)                // [B,K,D]
{
  // U in B-fragment order: [kstep(8)][ntile(16)][lane(32)][e(16)] bf16 = 128 KB
  __shared__ __bf16 U_frag[8 * 16 * 32 * 16];
  __shared__ __bf16 h_bf[16 * HBST];        // bf16 h, row-major (A fragments)
  __shared__ float  h_lds[16 * HSTR];       // fp32 h (gates / hold / writeback)
  __shared__ float  keys_lds[16 * HSTR];
  __shared__ float  kv_lds[16 * HSTR];
  __shared__ float  enc_buf[2][DD];         // TDM double buffers
  __shared__ float  encw_buf[2][DD];
  __shared__ float  gates[16];
  __shared__ float  norm[16];

  const int tid  = threadIdx.x;
  const int lane = tid & 31;
  const int wave = tid >> 5;
  const int half = lane >> 4;
  const int nlo  = lane & 15;
  const int b    = blockIdx.x >> 2;         // 4 entity tiles per batch element
  const int k0   = (blockIdx.x & 3) * 16;
  const int n0   = wave * 32;
  const int nt0  = wave * 2;                // this wave's two 16-col tiles

  // ---- prologue: swizzle U into fragment order; stage keys/keysV; zero h ----
  for (int i = tid; i < 8 * 16 * 32 * 16; i += 256) {
    const int e  = i & 15;
    const int l  = (i >> 4) & 31;
    const int nt = (i >> 9) & 15;
    const int ks = i >> 13;
    const int kb = ks * 32 + 8 * (l >> 4) + e + ((e & 8) ? 8 : 0);
    const int n  = nt * 16 + (l & 15);
    U_frag[i] = (__bf16)U[(size_t)kb * DD + n];
  }
  for (int i = tid; i < 16 * DD; i += 256) {
    const int m = i >> 8, d = i & 255;
    const size_t g = ((size_t)b * KK + k0 + m) * DD + d;
    h_lds[m * HSTR + d]    = 0.f;
    h_bf[m * HBST + d]     = (__bf16)0.f;
    keys_lds[m * HSTR + d] = keys[g];
    kv_lds[m * HSTR + d]   = keysV[g];
  }

  // ---- TDM: preload step 0 streams ----
  const size_t base = (size_t)b * SS * DD;
  if (wave == 0) {
    tdm_load_row(lds_offset(&enc_buf[0][0]),  enc  + base, DD);
    tdm_load_row(lds_offset(&encw_buf[0][0]), encW + base, DD);
  }
  __syncthreads();

  for (int t = 0; t < SS; ++t) {
    const int cur = t & 1;
    // ---- stage 0: wait current stream, kick off t+1, reset norms ----
    if (wave == 0) {
      __builtin_amdgcn_s_wait_tensorcnt(0);
      if (t + 1 < SS) {
        const size_t rg = base + (size_t)(t + 1) * DD;
        tdm_load_row(lds_offset(&enc_buf[cur ^ 1][0]),  enc  + rg, DD);
        tdm_load_row(lds_offset(&encw_buf[cur ^ 1][0]), encW + rg, DD);
      }
    }
    if (tid < 16) norm[tid] = 0.f;
    __syncthreads();

    const float* enc_lds  = enc_buf[cur];
    const float* encw_lds = encw_buf[cur];

    // ---- stage 1: gates[k] = sigmoid( enc . (h[k]+keys[k]) ) ----
    {
      const int row = tid >> 4;     // 16 groups of 16 threads, one row each
      const int c   = tid & 15;
      float s = 0.f;
      for (int d = c; d < DD; d += 16)
        s += enc_lds[d] * (h_lds[row * HSTR + d] + keys_lds[row * HSTR + d]);
      s += __shfl_xor(s, 1); s += __shfl_xor(s, 2);
      s += __shfl_xor(s, 4); s += __shfl_xor(s, 8);
      if (c == 0) gates[row] = 1.f / (1.f + __expf(-s));
    }
    __syncthreads();

    // ---- stage 2: hU = h_tile(16x256) @ U via vectorized LDS fragments ----
    v8f acc0 = {}, acc1 = {};
    const __bf16* arow = &h_bf[nlo * HBST];
#pragma unroll 2
    for (int kk = 0; kk < DD; kk += 32) {
      const v8bf lo = *(const v8bf*)(arow + kk + 8 * half);
      const v8bf hi = *(const v8bf*)(arow + kk + 16 + 8 * half);
      const v16bf a = __builtin_shufflevector(lo, hi,
          0, 1, 2, 3, 4, 5, 6, 7, 8, 9, 10, 11, 12, 13, 14, 15);
      const int ks = kk >> 5;
      const v16bf b0 = *(const v16bf*)&U_frag[(((ks * 16) + nt0) * 32 + lane) * 16];
      const v16bf b1 = *(const v16bf*)&U_frag[(((ks * 16) + nt0 + 1) * 32 + lane) * 16];
      acc0 = __builtin_amdgcn_wmma_f32_16x16x32_bf16(false, a, false, b0, (short)0, acc0, false, false);
      acc1 = __builtin_amdgcn_wmma_f32_16x16x32_bf16(false, a, false, b1, (short)0, acc1, false, false);
    }

    // ---- epilogue: upd = h + gate*relu(hU + keysV + encW); norm partials ----
    float upd0[8], upd1[8], hold0[8], hold1[8];
#pragma unroll
    for (int r = 0; r < 8; ++r) {
      const int m  = r + 8 * half;
      const int nA = n0 + nlo, nB = nA + 16;
      const float g = gates[m];
      float t0 = fmaxf(acc0[r] + kv_lds[m * HSTR + nA] + encw_lds[nA], 0.f);
      float t1 = fmaxf(acc1[r] + kv_lds[m * HSTR + nB] + encw_lds[nB], 0.f);
      hold0[r] = h_lds[m * HSTR + nA];
      hold1[r] = h_lds[m * HSTR + nB];
      upd0[r]  = hold0[r] + g * t0;
      upd1[r]  = hold1[r] + g * t1;
      float sq = upd0[r] * upd0[r] + upd1[r] * upd1[r];
      sq += __shfl_xor(sq, 1); sq += __shfl_xor(sq, 2);
      sq += __shfl_xor(sq, 4); sq += __shfl_xor(sq, 8);
      if (nlo == 0) atomicAdd(&norm[m], sq);   // ds_add_f32
    }
    __syncthreads();

    // ---- stage 3: L2-normalize, masked writeback into resident h (f32+bf16) ----
    {
      const bool msk = mask[(size_t)b * SS + t] != 0;
#pragma unroll
      for (int r = 0; r < 8; ++r) {
        const int m  = r + 8 * half;
        const int nA = n0 + nlo, nB = nA + 16;
        const float rn = rsqrtf(fmaxf(norm[m], 1e-12f));
        const float w0 = msk ? upd0[r] * rn : hold0[r];
        const float w1 = msk ? upd1[r] * rn : hold1[r];
        h_lds[m * HSTR + nA] = w0;
        h_lds[m * HSTR + nB] = w1;
        h_bf[m * HBST + nA]  = (__bf16)w0;
        h_bf[m * HBST + nB]  = (__bf16)w1;
      }
    }
    __syncthreads();
  }

  // ---- write final h tile ----
  for (int i = tid; i < 16 * DD; i += 256) {
    const int m = i >> 8, d = i & 255;
    out[((size_t)b * KK + k0 + m) * DD + d] = h_lds[m * HSTR + d];
  }
}

// ---------------------------------------------------------------------------
extern "C" void kernel_launch(void* const* d_in, const int* in_sizes, int n_in,
                              void* d_out, int out_size, void* d_ws, size_t ws_size,
                              hipStream_t stream) {
  const float*         enc  = (const float*)d_in[0];          // [B,S,D]
  const unsigned char* mask = (const unsigned char*)d_in[1];  // [B,S] bool
  const float*         keys = (const float*)d_in[2];          // [B,K,D]
  const float*         U    = (const float*)d_in[3];          // [D,D]
  const float*         V    = (const float*)d_in[4];          // [D,D]
  const float*         W    = (const float*)d_in[5];          // [D,D]
  float* out = (float*)d_out;                                 // [B,K,D]

  float* encW  = (float*)d_ws;                    // [B,S,D] = 16 MB
  float* keysV = encW + (size_t)BB * SS * DD;     // [B,K,D] =  8 MB

  // Time-invariant GEMMs hoisted out of the scan.
  gemm256_bf16<<<(BB * SS) / 16, 256, 0, stream>>>(enc,  W, encW);   // enc @ W for all t
  gemm256_bf16<<<(BB * KK) / 16, 256, 0, stream>>>(keys, V, keysV);  // keys @ V

  // Sequential scan with h and U resident in LDS, TDM-streamed enc/encW.
  entity_scan<<<BB * (KK / 16), 256, 0, stream>>>(enc, mask, keys, U, encW, keysV, out);
}
